// LorentzConv1d_21199958573236
// MI455X (gfx1250) — matrix operand order repaired and checked by reading, not compiled, not run on hardware
//
#include <hip/hip_runtime.h>
#include <math.h>

typedef _Float16 v16h  __attribute__((ext_vector_type(16)));
typedef float    v8f   __attribute__((ext_vector_type(8)));
typedef unsigned u32x4 __attribute__((ext_vector_type(4)));

namespace cfg {
constexpr int kBS   = 16;
constexpr int kL    = 16384;
constexpr int kCIn  = 65;
constexpr int kCOut = 65;
constexpr int kKsz  = 5;
constexpr int kPad  = 2;
constexpr int kTB   = 128;   // positions per block
constexpr int kK    = 320;   // space features (5*64), multiple of 32
constexpr int kNCh  = 64;    // useful output channels (o = 1..64)
constexpr int kLinF = kK + 1; // 321, W row stride
}

union FragU { u32x4 u[2]; v16h h; };
union H8    { u32x4 u; _Float16 f[8]; };

__global__ __launch_bounds__(256)
void lorentz_conv1d_wmma(const float* __restrict__ x,
                         const float* __restrict__ W,
                         const float* __restrict__ bias,
                         float* __restrict__ out)
{
  using namespace cfg;
  __shared__ _Float16 sX[(kTB + kKsz - 1) * kNCh]; // 132*64 f16 = 16896 B
  __shared__ _Float16 sB[kNCh * kK];               // 64*320 f16 = 40960 B
  __shared__ float    sT[kTB];                     // 512 B

  const int tid   = threadIdx.x;
  const int blk   = blockIdx.x;
  const int batch = blk >> 7;          // 128 blocks per batch row
  const int n0    = (blk & 127) * kTB;
  const float* xb = x + (size_t)batch * kL * kCIn;

  // ---- stage space channels (zero-padded at sequence edges) into LDS f16 ----
  for (int i = tid; i < (kTB + kKsz - 1) * kNCh; i += 256) {
    const int r = i >> 6, c = i & 63;
    const int n = n0 - kPad + r;
    float v = ((unsigned)n < (unsigned)kL) ? xb[(size_t)n * kCIn + 1 + c] : 0.f;
    sX[i] = (_Float16)v;
  }
  // ---- stage W (drop time row o=0 and time column f=0) into LDS f16 ----
  {
    const int o = tid >> 2;       // 0..63 -> channel o+1
    const int part = tid & 3;     // 80 columns each
    const float* wrow = W + (size_t)(o + 1) * kLinF + 1;
    for (int j = part * 80; j < part * 80 + 80; ++j)
      sB[o * kK + j] = (_Float16)wrow[j];
  }
  __syncthreads();

  // ---- per-position Lorentz time t = sqrt(1 + ||s||^2) (contiguous window) ----
  if (tid < kTB) {
    float s = 0.f;
    const _Float16* row = sX + tid * kNCh;
    for (int j = 0; j < kK; j += 8) {
      H8 hh; hh.u = *(const u32x4*)(row + j);
      #pragma unroll
      for (int e = 0; e < 8; ++e) { float v = (float)hh.f[e]; s += v * v; }
    }
    sT[tid] = sqrtf(1.f + s);
  }
  __syncthreads();

  // ---- WMMA GEMM: each wave = 16 positions x 64 channels, K = 320 ----
  const int wave = tid >> 5;
  const int lane = tid & 31;
  const int m    = lane & 15;   // A row / B,C column within tile
  const int hv   = lane >> 4;   // half-wave select
  const int pt   = wave;        // position tile 0..7

  v8f acc0 = {}, acc1 = {}, acc2 = {}, acc3 = {};
  const int aBase = (pt * 16 + m) * kNCh; // contiguous 320-feature window start

  #pragma unroll
  for (int kc = 0; kc < kK / 32; ++kc) {
    // A (16x32 f16): lanes 0-15 K=0..15, lanes 16-31 K=16..31
    FragU a;
    a.u[0] = *(const u32x4*)(sX + aBase + kc * 32 + hv * 16);
    a.u[1] = *(const u32x4*)(sX + aBase + kc * 32 + hv * 16 + 8);
    // B (32x16 f16): lanes 0-15 K=0..15, lanes 16-31 K=16..31 (col = m)
    const int kb = kc * 32 + hv * 16;
    FragU b0, b1, b2, b3;
    b0.u[0] = *(const u32x4*)(sB + (0 * 16 + m) * kK + kb);
    b0.u[1] = *(const u32x4*)(sB + (0 * 16 + m) * kK + kb + 8);
    b1.u[0] = *(const u32x4*)(sB + (1 * 16 + m) * kK + kb);
    b1.u[1] = *(const u32x4*)(sB + (1 * 16 + m) * kK + kb + 8);
    b2.u[0] = *(const u32x4*)(sB + (2 * 16 + m) * kK + kb);
    b2.u[1] = *(const u32x4*)(sB + (2 * 16 + m) * kK + kb + 8);
    b3.u[0] = *(const u32x4*)(sB + (3 * 16 + m) * kK + kb);
    b3.u[1] = *(const u32x4*)(sB + (3 * 16 + m) * kK + kb + 8);

    acc0 = __builtin_amdgcn_wmma_f32_16x16x32_f16(false, a.h, false, b0.h, (short)0, acc0, false, false);
    acc1 = __builtin_amdgcn_wmma_f32_16x16x32_f16(false, a.h, false, b1.h, (short)0, acc1, false, false);
    acc2 = __builtin_amdgcn_wmma_f32_16x16x32_f16(false, a.h, false, b2.h, (short)0, acc2, false, false);
    acc3 = __builtin_amdgcn_wmma_f32_16x16x32_f16(false, a.h, false, b3.h, (short)0, acc3, false, false);
  }

  // ---- epilogue: rank-1 time-column update + bias, re-add output time ----
  float w1v[4], bbv[4];
  #pragma unroll
  for (int nt = 0; nt < 4; ++nt) {
    const int ch = 1 + nt * 16 + m;
    w1v[nt] = W[(size_t)ch * kLinF];   // W[ch, 0] multiplies t
    bbv[nt] = bias[ch];
  }

  #pragma unroll
  for (int r = 0; r < 8; ++r) {
    const int pl = pt * 16 + r + 8 * hv;   // C layout: M = r + 8*(lane/16)
    const int n  = n0 + pl;
    const float t = sT[pl];
    float y0 = acc0[r] + t * w1v[0] + bbv[0];
    float y1 = acc1[r] + t * w1v[1] + bbv[1];
    float y2 = acc2[r] + t * w1v[2] + bbv[2];
    float y3 = acc3[r] + t * w1v[3] + bbv[3];
    float sq = y0 * y0 + y1 * y1 + y2 * y2 + y3 * y3;
    // reduce over the 16 channel-lanes of this half-wave
    sq += __shfl_xor(sq, 1, 32);
    sq += __shfl_xor(sq, 2, 32);
    sq += __shfl_xor(sq, 4, 32);
    sq += __shfl_xor(sq, 8, 32);
    const float tm = sqrtf(1.f + sq);
    float* row = out + ((size_t)batch * kL + n) * kCOut;
    row[1 + 0 * 16 + m] = y0;
    row[1 + 1 * 16 + m] = y1;
    row[1 + 2 * 16 + m] = y2;
    row[1 + 3 * 16 + m] = y3;
    if (m == 0) row[0] = tm;
  }
}

extern "C" void kernel_launch(void* const* d_in, const int* in_sizes, int n_in,
                              void* d_out, int out_size, void* d_ws, size_t ws_size,
                              hipStream_t stream) {
  (void)in_sizes; (void)n_in; (void)d_ws; (void)ws_size; (void)out_size;
  const float* x = (const float*)d_in[0];   // [16, 16384, 65] f32
  const float* W = (const float*)d_in[1];   // [65, 321] f32
  const float* b = (const float*)d_in[2];   // [65] f32
  float* out = (float*)d_out;               // [16, 16384, 65] f32
  const int blocks = cfg::kBS * (cfg::kL / cfg::kTB); // 2048
  lorentz_conv1d_wmma<<<blocks, 256, 0, stream>>>(x, W, b, out);
}
